// LocalMSA_7103875907871
// MI455X (gfx1250) — compile-verified, matrix operand order checked by reading, not run on hardware
//
#include <hip/hip_runtime.h>
#include <hip/hip_bf16.h>

typedef __attribute__((ext_vector_type(16))) _Float16     v16h;
typedef __attribute__((ext_vector_type(8)))  float        v8f;
typedef __attribute__((ext_vector_type(4)))  unsigned int v4u;

#define IMG   256
#define CDIM  128
#define NH    4
#define HD    32
#define WSZ   8
#define NTOK  64
#define NPOS  100   // 10x10 halo positions
#define PN    112   // padded to 7 WMMA n-tiles

union AF { v16h h; v4u q[2]; };

// Slot of K index kk (0..31) inside a 32-half fragment-major group:
// lanes 0-15 consume kk {0..7,16..23} (slots 0..15), lanes 16-31 kk {8..15,24..31}.
__device__ __forceinline__ int bslot(int kk) {
  int kke  = kk & 30;
  int hi   = (kke >> 3) & 1;
  int base = kke - 8 * hi;
  int v    = (base < 8) ? (base >> 1) : (4 + ((base - 16) >> 1));
  return hi * 16 + 2 * v + (kk & 1);
}

__device__ __forceinline__ v8f wmma_f16(const AF& a, const AF& b, v8f c) {
  return __builtin_amdgcn_wmma_f32_16x16x32_f16(false, a.h, false, b.h,
                                                (short)0, c, false, false);
}

// Load A fragment from row-major [row][K] f16: two 16B chunks per lane.
__device__ __forceinline__ AF load_a(const _Float16* row, int ks, int hi) {
  AF a;
  const _Float16* p = row + ks * 32 + hi * 8;
  a.q[0] = *reinterpret_cast<const v4u*>(p);
  a.q[1] = *reinterpret_cast<const v4u*>(p + 16);
  return a;
}

// Load B fragment from fragment-major [ks][n][32] f16: one 32B chunk per lane.
__device__ __forceinline__ AF load_b(const _Float16* grp, int hi) {
  AF b;
  const _Float16* p = grp + hi * 16;
  b.q[0] = *reinterpret_cast<const v4u*>(p);
  b.q[1] = *reinterpret_cast<const v4u*>(p + 8);
  return b;
}

// ---------------------------------------------------------------------------
// Kernel 1: var[b,y,x] = mean_c sqrt(sobel_x(xd)^2 + sobel_y(xd)^2 + 1e-6),
//           xd = depthwise3x3(x, w_x_dw). 16x16 pixel tile per block.
// ---------------------------------------------------------------------------
__global__ __launch_bounds__(256) void LocalMSA_var_kernel(
    const float* __restrict__ x, const float* __restrict__ w_x_dw,
    float* __restrict__ varbuf) {
  __shared__ float wdw[CDIM * 9];
  __shared__ float xs[20 * 20];
  __shared__ float xds[18 * 18];
  const int tid = threadIdx.x;
  const int b   = blockIdx.y;
  const int ty0 = (blockIdx.x >> 4) * 16;
  const int tx0 = (blockIdx.x & 15) * 16;
  const int py = tid >> 4, px = tid & 15;

  for (int i = tid; i < CDIM * 9; i += 256) wdw[i] = w_x_dw[i];

  float acc = 0.f;
  for (int c = 0; c < CDIM; ++c) {
    __syncthreads();
    for (int i = tid; i < 400; i += 256) {
      int ly = i / 20, lx = i - ly * 20;
      int gy = ty0 - 2 + ly, gx = tx0 - 2 + lx;
      xs[i] = (gy >= 0 && gy < IMG && gx >= 0 && gx < IMG)
            ? x[(((size_t)b * CDIM + c) * IMG + gy) * IMG + gx] : 0.f;
    }
    __syncthreads();
    for (int i = tid; i < 324; i += 256) {
      int dy = i / 18, dx = i - dy * 18;
      int gy = ty0 - 1 + dy, gx = tx0 - 1 + dx;
      float s = 0.f;
      if (gy >= 0 && gy < IMG && gx >= 0 && gx < IMG) {
        #pragma unroll
        for (int ky = 0; ky < 3; ++ky)
          #pragma unroll
          for (int kx = 0; kx < 3; ++kx)
            s += xs[(dy + ky) * 20 + dx + kx] * wdw[c * 9 + ky * 3 + kx];
      }
      xds[i] = s;
    }
    __syncthreads();
    float g_x = 0.f, g_y = 0.f;
    #pragma unroll
    for (int ky = 0; ky < 3; ++ky) {
      #pragma unroll
      for (int kx = 0; kx < 3; ++kx) {
        const float SX[3][3] = {{-1.f,0.f,1.f},{-2.f,0.f,2.f},{-1.f,0.f,1.f}};
        float v = xds[(py + ky) * 18 + px + kx];
        g_x += v * SX[ky][kx];
        g_y += v * SX[kx][ky];
      }
    }
    acc += sqrtf(g_x * g_x + g_y * g_y + 1e-6f);
  }
  varbuf[(size_t)b * IMG * IMG + (ty0 + py) * IMG + (tx0 + px)] =
      acc * (1.f / 128.f);
}

// ---------------------------------------------------------------------------
// Kernel 2: fully fused window pipeline (one 8x8 window per block, 8 waves).
//   qkv1x1 (WMMA) -> dw3x3 -> QK^T (WMMA) -> bias+softmax -> AV (WMMA)
//   -> proj (WMMA) + bias -> d_out
// All B operands stored fragment-major ([ks][n][32]) -> 2x ds_load_b128/frag.
// ---------------------------------------------------------------------------
__global__ __launch_bounds__(256) void LocalMSA_attn_kernel(
    const float* __restrict__ x,
    const float* __restrict__ w_qkv,
    const float* __restrict__ w_qkv_dw,
    const float* __restrict__ w_proj,
    const float* __restrict__ b_proj,
    const float* __restrict__ temperature,
    const float* __restrict__ pos_emb,
    const float* __restrict__ varbuf,
    float* __restrict__ out) {
  extern __shared__ __align__(16) _Float16 smem[];
  _Float16* xt   = smem;                  // B frag-major: [ks4][p PN][32]
  _Float16* aL   = xt + CDIM * PN;        // [128][128] row-major A operand
  _Float16* scr  = aL + CDIM * CDIM;      // qs [128][PN] / attn [4][64][64]
  _Float16* qwin = scr + 16384;           // [head][tok][hd] row-major (A op)
  _Float16* kwin = qwin + CDIM * NTOK;    // [head] frag-major [tok64][32]
  _Float16* vwin = kwin + CDIM * NTOK;    // [head] frag-major [ks2][hd32][32]
  _Float16* owin = qwin;                  // reuse: frag-major [ks4][tok64][32]
  __shared__ float varw[NTOK];

  const int tid  = threadIdx.x;
  const int lane = tid & 31;
  const int wave = tid >> 5;
  const int li = lane & 15;
  const int hi = lane >> 4;
  const int bb = blockIdx.y;
  const int wy = blockIdx.x >> 5;
  const int wx = blockIdx.x & 31;
  const int y0 = wy * WSZ - 1, x0 = wx * WSZ - 1;

  if (tid < NTOK)
    varw[tid] = varbuf[(size_t)bb * IMG * IMG +
                       (wy * WSZ + (tid >> 3)) * IMG + wx * WSZ + (tid & 7)];

  // ---- load x halo tile (10x10 x 128ch) as f16, fragment-major over K=c ----
  for (int idx = tid; idx < CDIM * NPOS; idx += 256) {
    int c = idx / NPOS, p = idx - c * NPOS;
    int py = p / 10, px = p - py * 10;
    int gy = y0 + py, gx = x0 + px;
    float v = (gy >= 0 && gy < IMG && gx >= 0 && gx < IMG)
            ? x[(((size_t)bb * CDIM + c) * IMG + gy) * IMG + gx] : 0.f;
    xt[((c >> 5) * PN + p) * 32 + bslot(c & 31)] = (_Float16)v;
  }
  for (int idx = tid; idx < CDIM * (PN - NPOS); idx += 256) {
    int c = idx / (PN - NPOS), p = NPOS + (idx - c * (PN - NPOS));
    xt[((c >> 5) * PN + p) * 32 + bslot(c & 31)] = (_Float16)0.f;
  }
  __syncthreads();

  // ---- 3 passes: q / k / v = dwconv3x3( W_qkv[s] @ x_halo ) ----
  for (int s = 0; s < 3; ++s) {
    for (int idx = tid; idx < CDIM * CDIM; idx += 256)
      aL[idx] = (_Float16)w_qkv[(size_t)s * CDIM * CDIM + idx];
    __syncthreads();

    const int m0 = wave * 16;
    const _Float16* arow = &aL[(m0 + li) * CDIM];
    for (int nt = 0; nt < 7; ++nt) {
      v8f acc = {};
      const int n = nt * 16 + li;
      for (int ks = 0; ks < 4; ++ks) {
        AF a  = load_a(arow, ks, hi);
        AF b2 = load_b(&xt[(ks * PN + n) * 32], hi);
        acc = wmma_f16(a, b2, acc);
      }
      #pragma unroll
      for (int r = 0; r < 8; ++r)
        scr[(m0 + r + hi * 8) * PN + n] = (_Float16)acc[r];
    }
    __syncthreads();

    const float qscale = (s == 0) ? 0.17677669529663687f : 1.0f; // hd^-0.5 on q
    for (int idx = tid; idx < CDIM * NTOK; idx += 256) {
      int c = idx >> 6, t = idx & 63;
      int py = (t >> 3) + 1, px = (t & 7) + 1;
      float sum = 0.f;
      #pragma unroll
      for (int ky = 0; ky < 3; ++ky)
        #pragma unroll
        for (int kx = 0; kx < 3; ++kx)
          sum += (float)scr[c * PN + (py + ky - 1) * 10 + (px + kx - 1)]
               * w_qkv_dw[((size_t)s * CDIM + c) * 9 + ky * 3 + kx];
      int h = c >> 5, kd = c & 31;
      _Float16 hv = (_Float16)(sum * qscale);
      if (s == 0)      qwin[h * 2048 + t * HD + kd] = hv;                       // A row-major
      else if (s == 1) kwin[h * 2048 + t * 32 + bslot(kd)] = hv;                // B: K=hd, n=tok
      else             vwin[h * 2048 + ((t >> 5) * HD + kd) * 32 + bslot(t & 31)] = hv; // B: K=tok, n=hd
    }
    __syncthreads();
  }

  // ---- attention: 2 waves per head, 2 token m-tiles per wave ----
  const int h  = wave >> 1;
  const int mg = wave & 1;
  const float temp = temperature[h];

  for (int mi = 0; mi < 2; ++mi) {
    const int m0 = (mg * 2 + mi) * 16;
    v8f st[4];
    v8f z = {};
    #pragma unroll
    for (int nt = 0; nt < 4; ++nt) st[nt] = z;
    #pragma unroll
    for (int nt = 0; nt < 4; ++nt) {        // sim = (q*scale) @ k^T, K = hd = 32
      AF a  = load_a(&qwin[h * 2048 + (m0 + li) * HD], 0, hi);
      AF b2 = load_b(&kwin[h * 2048 + (nt * 16 + li) * 32], hi);
      st[nt] = wmma_f16(a, b2, st[nt]);
    }
    // bias + row softmax (rows live within a 16-lane half in C layout)
    #pragma unroll
    for (int r = 0; r < 8; ++r) {
      const int i = m0 + r + hi * 8;
      const float vb = varw[i] * temp;
      float rowv[4], mx = -3.0e38f;
      #pragma unroll
      for (int nt = 0; nt < 4; ++nt) {
        int j = nt * 16 + li;
        float sv = st[nt][r] + pos_emb[((size_t)h * NTOK + i) * NTOK + j]
                 + (i == j ? vb : 0.f);
        rowv[nt] = sv;
        mx = fmaxf(mx, sv);
      }
      #pragma unroll
      for (int d = 1; d < 16; d <<= 1) mx = fmaxf(mx, __shfl_xor(mx, d, 32));
      float ssum = 0.f;
      #pragma unroll
      for (int nt = 0; nt < 4; ++nt) { rowv[nt] = __expf(rowv[nt] - mx); ssum += rowv[nt]; }
      #pragma unroll
      for (int d = 1; d < 16; d <<= 1) ssum += __shfl_xor(ssum, d, 32);
      float inv = 1.f / ssum;
      #pragma unroll
      for (int nt = 0; nt < 4; ++nt)
        scr[(h * NTOK + i) * NTOK + nt * 16 + li] = (_Float16)(rowv[nt] * inv);
    }
  }
  __syncthreads();

  // ---- out = attn @ V  (K = 64 tokens -> 2 k-steps), into owin frag-major ----
  for (int mi = 0; mi < 2; ++mi) {
    const int m0 = (mg * 2 + mi) * 16;
    const _Float16* arow = &scr[(h * NTOK + m0 + li) * NTOK];
    for (int nv = 0; nv < 2; ++nv) {
      v8f acc = {};
      #pragma unroll
      for (int ks = 0; ks < 2; ++ks) {
        AF a  = load_a(arow, ks, hi);
        AF b2 = load_b(&vwin[h * 2048 + (ks * HD + nv * 16 + li) * 32], hi);
        acc = wmma_f16(a, b2, acc);
      }
      #pragma unroll
      for (int r = 0; r < 8; ++r) {
        int tok = m0 + r + hi * 8;
        int c = h * HD + nv * 16 + li;             // proj K-dim channel
        owin[((c >> 5) * NTOK + tok) * 32 + bslot(c & 31)] = (_Float16)acc[r];
      }
    }
  }
  __syncthreads();

  // ---- proj: W_proj[128x128] @ owin[128x64] + bias -> d_out ----
  for (int idx = tid; idx < CDIM * CDIM; idx += 256)
    aL[idx] = (_Float16)w_proj[idx];
  __syncthreads();

  const int m0p = wave * 16;
  const _Float16* arow = &aL[(m0p + li) * CDIM];
  for (int nt = 0; nt < 4; ++nt) {
    v8f acc = {};
    for (int ks = 0; ks < 4; ++ks) {
      AF a  = load_a(arow, ks, hi);
      AF b2 = load_b(&owin[(ks * NTOK + nt * 16 + li) * 32], hi);
      acc = wmma_f16(a, b2, acc);
    }
    #pragma unroll
    for (int r = 0; r < 8; ++r) {
      int co  = m0p + r + hi * 8;
      int tok = nt * 16 + li;
      int yy = wy * WSZ + (tok >> 3), xx = wx * WSZ + (tok & 7);
      out[(((size_t)bb * CDIM + co) * IMG + yy) * IMG + xx] = acc[r] + b_proj[co];
    }
  }
}

#define SMEM_BYTES ((CDIM * PN + CDIM * CDIM + 16384 + 3 * CDIM * NTOK) * 2)

extern "C" void kernel_launch(void* const* d_in, const int* in_sizes, int n_in,
                              void* d_out, int out_size, void* d_ws, size_t ws_size,
                              hipStream_t stream) {
  const float* x           = (const float*)d_in[0];
  const float* w_qkv       = (const float*)d_in[1];
  const float* w_qkv_dw    = (const float*)d_in[2];
  const float* w_x_dw      = (const float*)d_in[3];
  const float* w_proj      = (const float*)d_in[4];
  const float* b_proj      = (const float*)d_in[5];
  const float* temperature = (const float*)d_in[6];
  const float* pos_emb     = (const float*)d_in[7];
  float* out    = (float*)d_out;
  float* varbuf = (float*)d_ws;   // 4*256*256 f32 = 1 MiB

  // Edge-statistic map (Sobel on depthwise-convolved x)
  LocalMSA_var_kernel<<<dim3(256, 4), 256, 0, stream>>>(x, w_x_dw, varbuf);

  // Fused qkv -> window attention -> projection, one block per 8x8 window
  LocalMSA_attn_kernel<<<dim3(1024, 4), 256, SMEM_BYTES, stream>>>(
      x, w_qkv, w_qkv_dw, w_proj, b_proj, temperature, pos_emb, varbuf, out);
}
